// SRNN_66632122630367
// MI455X (gfx1250) — compile-verified, matrix operand order checked by reading, not run on hardware
//
#include <hip/hip_runtime.h>
#include <math.h>

// ---------------------------------------------------------------------------
// Types for CDNA5 WMMA (wave32, V_WMMA_F32_16X16X32_BF16)
// ---------------------------------------------------------------------------
typedef __attribute__((ext_vector_type(16))) __bf16 v16bf;
typedef __attribute__((ext_vector_type(8)))  float  v8f;

struct U16x16 { unsigned short v[16]; };

__device__ __forceinline__ unsigned short f2bf(float f) {
    unsigned int u = __float_as_uint(f);
    unsigned int r = u + 0x7FFFu + ((u >> 16) & 1u);   // round-to-nearest-even
    return (unsigned short)(r >> 16);
}
__device__ __forceinline__ float bf2f(unsigned short h) {
    return __uint_as_float(((unsigned int)h) << 16);
}
__device__ __forceinline__ float sigmoidf(float x) { return 1.0f / (1.0f + expf(-x)); }

// ---------------------------------------------------------------------------
// Problem constants
// ---------------------------------------------------------------------------
#define BB 64          // batch
#define TT 64          // time steps
#define NN 8           // nodes
#define EE 256         // edge/node feature dim
#define ROWS 512       // B*N
#define MROWS 4096     // ROWS + ROWS*7 (attention GEMM rows)

// ---------------------------------------------------------------------------
// Generic bf16 WMMA GEMM:  out = act( A[MxK] * W[NxK]^T + bias )
//   A, W row-major bf16 (ushort). M%64==0, N%64==0, K%64==0.
//   128 threads = 4 waves; block tile 64x64; each wave a 32x32 sub-tile.
//   K staged in 64-wide chunks through LDS.  Prefetch of chunk k+1 is issued
//   AFTER the post-store barrier (named scalar float4 regs, clamped offset on
//   the last chunk), so the VMEM latency overlaps the 8 WMMAs of the current
//   chunk; the only forced loadcnt-0 wait is at the next iteration's ds_store.
// ---------------------------------------------------------------------------
__global__ __launch_bounds__(128) void gemm_bf16_wmma(
    const unsigned short* __restrict__ A, int lda,
    const unsigned short* __restrict__ W, int ldw,
    const float* __restrict__ bias,
    float* __restrict__ outF, int ldof,
    unsigned short* __restrict__ outB, int ldob,
    int M, int N, int K, int relu)
{
    __shared__ __align__(16) unsigned short sA[64][64];
    __shared__ __align__(16) unsigned short sB[64][64];

    const int tid  = threadIdx.x;
    const int wave = tid >> 5;
    const int lane = tid & 31;
    const int m0 = blockIdx.y * 64;
    const int n0 = blockIdx.x * 64;
    const int wm = (wave >> 1) * 32;
    const int wn = (wave & 1) * 32;

    v8f acc[2][2] = {};

    const int fr = lane & 15;
    const int kb = (lane >> 4) * 8;

    // Per-thread copy coordinates: 4096 elems / 128 thr = 32 each as 4x float4.
    // chunk c covers rows c*16 + tid/8, column (tid&7)*8.
    const int rr = tid >> 3;            // 0..15
    const int cc = (tid & 7) * 8;       // 0,8,...,56
    const unsigned short* gA = A + (size_t)(m0 + rr) * lda + cc;
    const unsigned short* gW = W + (size_t)(n0 + rr) * ldw + cc;
    const size_t sa16 = (size_t)16 * lda;
    const size_t sw16 = (size_t)16 * ldw;

    float4 a0, a1, a2, a3, b0, b1, b2, b3;

#define LOAD_CHUNK(k0)                                          \
    do {                                                        \
        a0 = *(const float4*)(gA + (k0));                       \
        a1 = *(const float4*)(gA + sa16 + (k0));                \
        a2 = *(const float4*)(gA + 2 * sa16 + (k0));            \
        a3 = *(const float4*)(gA + 3 * sa16 + (k0));            \
        b0 = *(const float4*)(gW + (k0));                       \
        b1 = *(const float4*)(gW + sw16 + (k0));                \
        b2 = *(const float4*)(gW + 2 * sw16 + (k0));            \
        b3 = *(const float4*)(gW + 3 * sw16 + (k0));            \
    } while (0)

#define STORE_CHUNK()                                           \
    do {                                                        \
        *(float4*)(&sA[rr][cc])      = a0;                      \
        *(float4*)(&sA[rr + 16][cc]) = a1;                      \
        *(float4*)(&sA[rr + 32][cc]) = a2;                      \
        *(float4*)(&sA[rr + 48][cc]) = a3;                      \
        *(float4*)(&sB[rr][cc])      = b0;                      \
        *(float4*)(&sB[rr + 16][cc]) = b1;                      \
        *(float4*)(&sB[rr + 32][cc]) = b2;                      \
        *(float4*)(&sB[rr + 48][cc]) = b3;                      \
    } while (0)

    LOAD_CHUNK(0);
    for (int k0 = 0; k0 < K; k0 += 64) {
        __syncthreads();                 // previous chunk's compute done
        STORE_CHUNK();
        __syncthreads();                 // tiles visible to all waves
        int kn = (k0 + 64 < K) ? (k0 + 64) : k0;   // clamped -> unconditional
        LOAD_CHUNK(kn);                  // issued after barrier: overlaps WMMAs

#pragma unroll
        for (int ks = 0; ks < 64; ks += 32) {
            v16bf aV[2], bV[2];
#pragma unroll
            for (int t = 0; t < 2; ++t) {
                U16x16 at, bt;
                const int am = wm + t * 16 + fr;
                const int bn = wn + t * 16 + fr;
#pragma unroll
                for (int j = 0; j < 8; ++j) {
                    int kk = ks + kb + ((j < 4) ? (2 * j) : (16 + 2 * (j - 4)));
                    at.v[2 * j]     = sA[am][kk];
                    at.v[2 * j + 1] = sA[am][kk + 1];
                    bt.v[2 * j]     = sB[bn][kk];
                    bt.v[2 * j + 1] = sB[bn][kk + 1];
                }
                aV[t] = __builtin_bit_cast(v16bf, at);
                bV[t] = __builtin_bit_cast(v16bf, bt);
            }
#pragma unroll
            for (int i = 0; i < 2; ++i)
#pragma unroll
                for (int j = 0; j < 2; ++j)
                    acc[i][j] = __builtin_amdgcn_wmma_f32_16x16x32_bf16(
                        false, aV[i], false, bV[j], (short)0, acc[i][j], false, false);
        }
    }
#undef LOAD_CHUNK
#undef STORE_CHUNK

    // Epilogue.  D layout: lane col = lane&15, rows p + (lane<16 ? 0 : 8)
    const int cn_   = lane & 15;
    const int rbase = (lane >> 4) * 8;
#pragma unroll
    for (int i = 0; i < 2; ++i)
#pragma unroll
        for (int j = 0; j < 2; ++j)
#pragma unroll
            for (int p = 0; p < 8; ++p) {
                int m = m0 + wm + i * 16 + rbase + p;
                int n = n0 + wn + j * 16 + cn_;
                float v = acc[i][j][p];
                if (bias) v += bias[n];
                if (relu) v = fmaxf(v, 0.0f);
                if (outF) outF[(size_t)m * ldof + n] = v;
                if (outB) outB[(size_t)m * ldob + n] = f2bf(v);
            }
}

// ---------------------------------------------------------------------------
// Weight prep kernels (fp32 -> bf16, with concatenation along K)
// ---------------------------------------------------------------------------
__global__ void pack_bf16(const float* __restrict__ src,
                          unsigned short* __restrict__ dst, int n) {
    for (int i = blockIdx.x * blockDim.x + threadIdx.x; i < n;
         i += gridDim.x * blockDim.x)
        dst[i] = f2bf(src[i]);
}

__global__ void pack_cat(const float* __restrict__ s1, int c1,
                         const float* __restrict__ s2, int c2,
                         unsigned short* __restrict__ dst, int rows) {
    int ctot = c1 + c2;
    int total = rows * ctot;
    for (int i = blockIdx.x * blockDim.x + threadIdx.x; i < total;
         i += gridDim.x * blockDim.x) {
        int r = i / ctot, c = i - r * ctot;
        float v = (c < c1) ? s1[(size_t)r * c1 + c] : s2[(size_t)r * c2 + (c - c1)];
        dst[i] = f2bf(v);
    }
}

__global__ void add_bias_vec(const float* __restrict__ a,
                             const float* __restrict__ b,
                             float* __restrict__ d, int n) {
    for (int i = blockIdx.x * blockDim.x + threadIdx.x; i < n;
         i += gridDim.x * blockDim.x)
        d[i] = a[i] + b[i];
}

// ---------------------------------------------------------------------------
// Per-step input gather+cast:
//   Abuf rows [0,512)     = tef[:, s]              (b*8+n)
//   Abuf rows [512,4096)  = gathered spatial edges at t=s+1
//   Xn   rows [0,512)     = node_features[:, s+1]
// ---------------------------------------------------------------------------
__global__ void cast_step(const float* __restrict__ nf,
                          const float* __restrict__ sef,
                          const float* __restrict__ tef,
                          unsigned short* __restrict__ Abuf,
                          unsigned short* __restrict__ Xn, int s) {
    const int total = MROWS * EE + ROWS * EE;
    for (int i = blockIdx.x * blockDim.x + threadIdx.x; i < total;
         i += gridDim.x * blockDim.x) {
        if (i < ROWS * EE) {
            int row = i >> 8, e = i & 255;
            int b = row >> 3, n = row & 7;
            Abuf[i] = f2bf(tef[(((size_t)b * TT + s) * NN + n) * EE + e]);
        } else if (i < MROWS * EE) {
            int j = i - ROWS * EE;
            int row = j >> 8, e = j & 255;
            int bn = row / 7, k = row - bn * 7;
            int b = bn >> 3, n = bn & 7;
            int jj = (k < n) ? k : k + 1;          // idx[n][k] = n*7 + jj
            int er = n * 7 + jj;
            Abuf[i] = f2bf(sef[(((size_t)b * TT + (s + 1)) * (NN * (NN - 1)) + er) * EE + e]);
        } else {
            int j = i - MROWS * EE;
            int row = j >> 8, f = j & 255;
            int b = row >> 3, n = row & 7;
            Xn[j] = f2bf(nf[(((size_t)b * TT + (s + 1)) * NN + n) * EE + f]);
        }
    }
}

// ---------------------------------------------------------------------------
// Attention: e_k = <m_i, m_j_k>;  w = e/sum(e);  edge_all = sum_k w_k * edge_k
// One block per (b,n); 256 threads (one per feature).
// ---------------------------------------------------------------------------
__global__ void attn_kernel(const float* __restrict__ Mbuf,
                            const unsigned short* __restrict__ Abuf,
                            unsigned short* __restrict__ EA) {
    int bn = blockIdx.x;
    int t = threadIdx.x;
    __shared__ float red[256];
    __shared__ float wk[8];

    const float* mi = Mbuf + (size_t)bn * EE;
    for (int k = 0; k < 7; ++k) {
        const float* mj = Mbuf + (size_t)(ROWS + bn * 7 + k) * EE;
        red[t] = mi[t] * mj[t];
        __syncthreads();
        for (int off = 128; off; off >>= 1) {
            if (t < off) red[t] += red[t + off];
            __syncthreads();
        }
        if (t == 0) wk[k] = red[0];
        __syncthreads();
    }
    if (t == 0) {
        float s = 0.f;
        for (int k = 0; k < 7; ++k) s += wk[k];
        float inv = 1.0f / s;
        for (int k = 0; k < 7; ++k) wk[k] *= inv;
    }
    __syncthreads();
    float acc = 0.f;
    for (int k = 0; k < 7; ++k)
        acc += wk[k] * bf2f(Abuf[(size_t)(ROWS + bn * 7 + k) * EE + t]);
    EA[(size_t)bn * EE + t] = f2bf(acc);
}

// ---------------------------------------------------------------------------
// LSTM elementwise update.  G: 512x1024 gates (i,f,g,o blocks of 256).
// Writes c (f32 state) and h (bf16) into up to two concat destinations.
// ---------------------------------------------------------------------------
__global__ void lstm_update(const float* __restrict__ G,
                            float* __restrict__ c,
                            unsigned short* __restrict__ h1, int ldh1,
                            unsigned short* __restrict__ h2, int ldh2) {
    int r = blockIdx.x;      // 0..511
    int j = threadIdx.x;     // 0..255
    const float* g = G + (size_t)r * 1024;
    float gi = sigmoidf(g[j]);
    float gf = sigmoidf(g[256 + j]);
    float gg = tanhf(g[512 + j]);
    float go = sigmoidf(g[768 + j]);
    float c2 = gf * c[(size_t)r * 256 + j] + gi * gg;
    c[(size_t)r * 256 + j] = c2;
    unsigned short h = f2bf(go * tanhf(c2));
    h1[(size_t)r * ldh1 + j] = h;
    if (h2) h2[(size_t)r * ldh2 + j] = h;
}

// ---------------------------------------------------------------------------
// Final head: sum over nodes, batchnorm over batch, classifier, softmax
// ---------------------------------------------------------------------------
__global__ void reduce_nodes(const float* __restrict__ OutBuf,
                             float* __restrict__ NodeOut) {
    int b = blockIdx.x, o = threadIdx.x;
    float a = 0.f;
    for (int n = 0; n < NN; ++n)
        a += OutBuf[(size_t)((b << 3) + n) * 256 + o];
    NodeOut[(size_t)b * 256 + o] = a;
}

__global__ void finalize(const float* __restrict__ NodeOut,
                         const float* __restrict__ gamma,
                         const float* __restrict__ beta,
                         const float* __restrict__ cls_w,
                         const float* __restrict__ cls_b,
                         float* __restrict__ ybuf,
                         float* __restrict__ out) {
    int t = threadIdx.x;  // 256 threads, one per feature
    float mean = 0.f;
    for (int b = 0; b < BB; ++b) mean += NodeOut[(size_t)b * 256 + t];
    mean *= (1.0f / BB);
    float var = 0.f;
    for (int b = 0; b < BB; ++b) {
        float d = NodeOut[(size_t)b * 256 + t] - mean;
        var += d * d;
    }
    var *= (1.0f / BB);
    float inv = rsqrtf(var + 1e-5f);
    for (int b = 0; b < BB; ++b)
        ybuf[(size_t)b * 256 + t] =
            gamma[t] * (NodeOut[(size_t)b * 256 + t] - mean) * inv + beta[t];
    __syncthreads();
    if (t < BB) {
        float logits[12];
        float mx = -1e30f;
        for (int o = 0; o < 12; ++o) {
            float a = cls_b[o];
            for (int k = 0; k < 256; ++k)
                a += ybuf[(size_t)t * 256 + k] * cls_w[(size_t)o * 256 + k];
            logits[o] = a;
            mx = fmaxf(mx, a);
        }
        float s = 0.f;
        for (int o = 0; o < 12; ++o) { logits[o] = expf(logits[o] - mx); s += logits[o]; }
        float invs = 1.0f / s;
        for (int o = 0; o < 12; ++o) out[(size_t)t * 12 + o] = logits[o] * invs;
    }
}

// ---------------------------------------------------------------------------
// Host orchestration
// ---------------------------------------------------------------------------
static inline size_t align256(size_t x) { return (x + 255) & ~(size_t)255; }

static void launch_gemm(hipStream_t st, const unsigned short* A, int lda,
                        const unsigned short* W, int ldw, const float* bias,
                        float* outF, int ldof, unsigned short* outB, int ldob,
                        int M, int N, int K, int relu) {
    dim3 g(N / 64, M / 64), b(128);
    gemm_bf16_wmma<<<g, b, 0, st>>>(A, lda, W, ldw, bias, outF, ldof, outB, ldob,
                                    M, N, K, relu);
}

extern "C" void kernel_launch(void* const* d_in, const int* in_sizes, int n_in,
                              void* d_out, int out_size, void* d_ws, size_t ws_size,
                              hipStream_t stream) {
    (void)in_sizes; (void)n_in; (void)out_size; (void)ws_size;

    const float* nf    = (const float*)d_in[2];   // (B,T,N,F)
    const float* sef   = (const float*)d_in[3];   // (B,T,56,E)
    const float* tef   = (const float*)d_in[4];   // (B,T,N,E)
    const float* att_w = (const float*)d_in[5];
    const float* att_b = (const float*)d_in[6];
    const float* t_emb_w = (const float*)d_in[7];
    const float* t_emb_b = (const float*)d_in[8];
    const float* t_wih = (const float*)d_in[9];
    const float* t_whh = (const float*)d_in[10];
    const float* t_bih = (const float*)d_in[11];
    const float* t_bhh = (const float*)d_in[12];
    const float* s_emb_w = (const float*)d_in[13];
    const float* s_emb_b = (const float*)d_in[14];
    const float* s_wih = (const float*)d_in[15];
    const float* s_whh = (const float*)d_in[16];
    const float* s_bih = (const float*)d_in[17];
    const float* s_bhh = (const float*)d_in[18];
    const float* n_emb_w = (const float*)d_in[19];
    const float* n_emb_b = (const float*)d_in[20];
    const float* ne_emb_w = (const float*)d_in[21];
    const float* ne_emb_b = (const float*)d_in[22];
    const float* n_wih = (const float*)d_in[23];
    const float* n_whh = (const float*)d_in[24];
    const float* n_bih = (const float*)d_in[25];
    const float* n_bhh = (const float*)d_in[26];
    const float* n_out_w = (const float*)d_in[27];
    const float* n_out_b = (const float*)d_in[28];
    const float* bn_gamma = (const float*)d_in[29];
    const float* bn_beta  = (const float*)d_in[30];
    const float* cls_w    = (const float*)d_in[31];
    const float* cls_b    = (const float*)d_in[32];

    // ---- workspace carve-up -------------------------------------------------
    char* base = (char*)d_ws;
    size_t off = 0;
    auto alloc = [&](size_t bytes) -> void* {
        void* p = base + off;
        off = align256(off + bytes);
        return p;
    };
    unsigned short* W_att  = (unsigned short*)alloc(256 * 256 * 2);
    unsigned short* W_temb = (unsigned short*)alloc(256 * 256 * 2);
    unsigned short* W_semb = (unsigned short*)alloc(256 * 256 * 2);
    unsigned short* W_nemb = (unsigned short*)alloc(256 * 256 * 2);
    unsigned short* W_ne   = (unsigned short*)alloc(256 * 512 * 2);
    unsigned short* W_tcat = (unsigned short*)alloc(1024 * 512 * 2);
    unsigned short* W_scat = (unsigned short*)alloc(1024 * 512 * 2);
    unsigned short* W_ncat = (unsigned short*)alloc(1024 * 768 * 2);
    unsigned short* W_nout = (unsigned short*)alloc(256 * 256 * 2);
    float* bt = (float*)alloc(1024 * 4);
    float* bs = (float*)alloc(1024 * 4);
    float* bn = (float*)alloc(1024 * 4);
    unsigned short* Abuf = (unsigned short*)alloc((size_t)MROWS * EE * 2);
    unsigned short* Xn   = (unsigned short*)alloc((size_t)ROWS * EE * 2);
    float*          Mbuf = (float*)alloc((size_t)MROWS * EE * 4);
    unsigned short* EA   = (unsigned short*)alloc((size_t)ROWS * EE * 2);
    unsigned short* TS_IN = (unsigned short*)alloc((size_t)ROWS * 512 * 2);  // [xt|ht]
    unsigned short* SS_IN = (unsigned short*)alloc((size_t)ROWS * 512 * 2);  // [xs|hs]
    unsigned short* NE_IN = (unsigned short*)alloc((size_t)ROWS * 512 * 2);  // [ht|hs]
    unsigned short* NIN   = (unsigned short*)alloc((size_t)ROWS * 768 * 2);  // [xn|xne|hn]
    float* G  = (float*)alloc((size_t)ROWS * 1024 * 4);
    float* ct = (float*)alloc((size_t)ROWS * 256 * 4);
    float* cs = (float*)alloc((size_t)ROWS * 256 * 4);
    float* cn = (float*)alloc((size_t)ROWS * 256 * 4);
    float* OutBuf  = (float*)alloc((size_t)ROWS * 256 * 4);
    float* NodeOut = (float*)alloc((size_t)BB * 256 * 4);
    float* Ybuf    = (float*)alloc((size_t)BB * 256 * 4);

    // ---- zero persistent state (h and c start at 0) -------------------------
    hipMemsetAsync(TS_IN, 0, (size_t)ROWS * 512 * 2, stream);
    hipMemsetAsync(SS_IN, 0, (size_t)ROWS * 512 * 2, stream);
    hipMemsetAsync(NE_IN, 0, (size_t)ROWS * 512 * 2, stream);
    hipMemsetAsync(NIN,   0, (size_t)ROWS * 768 * 2, stream);
    hipMemsetAsync(ct, 0, (size_t)ROWS * 256 * 4, stream);
    hipMemsetAsync(cs, 0, (size_t)ROWS * 256 * 4, stream);
    hipMemsetAsync(cn, 0, (size_t)ROWS * 256 * 4, stream);

    // ---- weight prep (fp32 -> bf16, concat) ---------------------------------
    auto pack = [&](const float* s, unsigned short* d, int n) {
        pack_bf16<<<(n + 255) / 256, 256, 0, stream>>>(s, d, n);
    };
    pack(att_w,   W_att,  256 * 256);
    pack(t_emb_w, W_temb, 256 * 256);
    pack(s_emb_w, W_semb, 256 * 256);
    pack(n_emb_w, W_nemb, 256 * 256);
    pack(ne_emb_w, W_ne,  256 * 512);
    pack(n_out_w, W_nout, 256 * 256);
    pack_cat<<<(1024 * 512 + 255) / 256, 256, 0, stream>>>(t_wih, 256, t_whh, 256, W_tcat, 1024);
    pack_cat<<<(1024 * 512 + 255) / 256, 256, 0, stream>>>(s_wih, 256, s_whh, 256, W_scat, 1024);
    pack_cat<<<(1024 * 768 + 255) / 256, 256, 0, stream>>>(n_wih, 512, n_whh, 256, W_ncat, 1024);
    add_bias_vec<<<4, 256, 0, stream>>>(t_bih, t_bhh, bt, 1024);
    add_bias_vec<<<4, 256, 0, stream>>>(s_bih, s_bhh, bs, 1024);
    add_bias_vec<<<4, 256, 0, stream>>>(n_bih, n_bhh, bn, 1024);

    // ---- recurrent time loop (63 steps) -------------------------------------
    for (int s = 0; s < TT - 1; ++s) {
        cast_step<<<1024, 256, 0, stream>>>(nf, sef, tef, Abuf, Xn, s);

        // m = [tef ; edges] @ att_w^T + att_b   (4096 x 256, K=256)
        launch_gemm(stream, Abuf, 256, W_att, 256, att_b, Mbuf, 256,
                    nullptr, 0, MROWS, 256, 256, 0);

        attn_kernel<<<ROWS, 256, 0, stream>>>(Mbuf, Abuf, EA);

        // embeddings (relu), write bf16 into concat slots
        launch_gemm(stream, Abuf, 256, W_temb, 256, t_emb_b, nullptr, 0,
                    TS_IN, 512, ROWS, 256, 256, 1);                 // xt
        launch_gemm(stream, EA, 256, W_semb, 256, s_emb_b, nullptr, 0,
                    SS_IN, 512, ROWS, 256, 256, 1);                 // xs
        launch_gemm(stream, Xn, 256, W_nemb, 256, n_emb_b, nullptr, 0,
                    NIN, 768, ROWS, 256, 256, 1);                   // xn

        // temporal LSTM
        launch_gemm(stream, TS_IN, 512, W_tcat, 512, bt, G, 1024,
                    nullptr, 0, ROWS, 1024, 512, 0);
        lstm_update<<<ROWS, 256, 0, stream>>>(G, ct, TS_IN + 256, 512, NE_IN, 512);

        // spatial LSTM
        launch_gemm(stream, SS_IN, 512, W_scat, 512, bs, G, 1024,
                    nullptr, 0, ROWS, 1024, 512, 0);
        lstm_update<<<ROWS, 256, 0, stream>>>(G, cs, SS_IN + 256, 512, NE_IN + 256, 512);

        // node-edge embedding (relu) from [ht|hs]
        launch_gemm(stream, NE_IN, 512, W_ne, 512, ne_emb_b, nullptr, 0,
                    NIN + 256, 768, ROWS, 256, 512, 1);             // xne

        // node LSTM: [xn|xne|hn] @ [n_wih|n_whh]^T
        launch_gemm(stream, NIN, 768, W_ncat, 768, bn, G, 1024,
                    nullptr, 0, ROWS, 1024, 768, 0);
        lstm_update<<<ROWS, 256, 0, stream>>>(G, cn, NIN + 512, 768, nullptr, 0);
    }

    // ---- head: out = hn @ n_out_w^T + b, sum nodes, BN, classifier, softmax --
    launch_gemm(stream, NIN + 512, 768, W_nout, 256, n_out_b, OutBuf, 256,
                nullptr, 0, ROWS, 256, 256, 0);
    reduce_nodes<<<BB, 256, 0, stream>>>(OutBuf, NodeOut);
    finalize<<<1, 256, 0, stream>>>(NodeOut, bn_gamma, bn_beta, cls_w, cls_b,
                                    Ybuf, (float*)d_out);
}
